// JetGraphClassifier_60447369724691
// MI455X (gfx1250) — matrix operand order repaired and checked by reading, not compiled, not run on hardware
//
#include <hip/hip_runtime.h>

#define NNODES 50000
#define NEDGES 800000
#define NGRAPH 512
#define BN_EPS 1e-5f

typedef __attribute__((ext_vector_type(16))) _Float16 v16h;
typedef __attribute__((ext_vector_type(8)))  _Float16 v8h;
typedef __attribute__((ext_vector_type(8)))  float    v8f;

// ---------------------------------------------------------------------------
// f32 -> f16 staging with K zero-padded to Kp (multiple of 32), row-major.
// ---------------------------------------------------------------------------
__global__ void k_cvt_a(const float* __restrict__ A, _Float16* __restrict__ A16,
                        int N, int K, int Kp) {
    long t = (long)blockIdx.x * blockDim.x + threadIdx.x;
    if (t >= (long)N * Kp) return;
    int row = (int)(t / Kp);
    int k   = (int)(t - (long)row * Kp);
    A16[t] = (_Float16)((k < K) ? A[(size_t)row * K + k] : 0.f);
}

// W[K x M] (row-major f32) -> WT16[M x Kp] (f16, transposed, zero-padded)
__global__ void k_cvt_wt(const float* __restrict__ W, _Float16* __restrict__ WT,
                         int K, int M, int Kp) {
    int t = blockIdx.x * blockDim.x + threadIdx.x;
    if (t >= M * Kp) return;
    int mcol = t / Kp;
    int k    = t - mcol * Kp;
    WT[t] = (_Float16)((k < K) ? W[(size_t)k * M + mcol] : 0.f);
}

// ---------------------------------------------------------------------------
// WMMA GEMM: C[N x (MT*16)] = A16[N x KP] @ WT16[(MT*16) x KP]^T  (+ bias)
// One wave computes a 16 x (MT*16) row-strip: A fragment reused across MT
// v_wmma_f32_16x16x32_f16 per K-step. All fragment loads are unconditional
// 16B/32B vector loads (no divergence; EXEC stays all-ones per wave).
// ---------------------------------------------------------------------------
template <int KP, int MT>
__global__ void k_wmma_gemm(const _Float16* __restrict__ A16,
                            const _Float16* __restrict__ WT16,
                            const float* __restrict__ bias,
                            float* __restrict__ C, int N, int add_bias)
{
    const int lane = threadIdx.x & 31;
    const int wave = threadIdx.x >> 5;
    const int rowTiles = N >> 4;
    const int rt = blockIdx.x * (blockDim.x >> 5) + wave;
    if (rt >= rowTiles) return;                    // wave-uniform guard
    const int row0 = rt << 4;
    const int m  = lane & 15;
    const int hi = lane >> 4;
    const int M  = MT * 16;

    // A fragment lanes: row m; K = hi*16 + {0..15}
    const _Float16* arow = A16 + (size_t)(row0 + m) * KP + hi * 16;

    v8f acc[MT];
    #pragma unroll
    for (int c = 0; c < MT; ++c) acc[c] = (v8f){};

    for (int k0 = 0; k0 < KP; k0 += 32) {
        v8h a0 = *(const v8h*)(arow + k0);         // 16B load
        v8h a1 = *(const v8h*)(arow + k0 + 8);     // 16B load
        v16h a = __builtin_shufflevector(a0, a1, 0, 1, 2, 3, 4, 5, 6, 7,
                                                 8, 9, 10, 11, 12, 13, 14, 15);
        #pragma unroll
        for (int c = 0; c < MT; ++c) {
            // B fragment lanes: column c*16+m; K = hi*16 + {0..15}
            v16h b = *(const v16h*)(WT16 + (size_t)(c * 16 + m) * KP + hi * 16 + k0);
            acc[c] = __builtin_amdgcn_wmma_f32_16x16x32_f16(
                false, a, false, b, (short)0, acc[c], false, false);
        }
    }

    #pragma unroll
    for (int c = 0; c < MT; ++c) {
        float bv = add_bias ? bias[c * 16 + m] : 0.f;
        #pragma unroll
        for (int v = 0; v < 8; ++v) {              // C/D layout: M = v + 8*hi
            C[(size_t)(row0 + v + 8 * hi) * M + c * 16 + m] = acc[c][v] + bv;
        }
    }
}

// ---------------------------------------------------------------------------
// GCN normalization: deg (self-loop weight 1) -> d^{-1/2}
// ---------------------------------------------------------------------------
__global__ void k_deg_init(float* deg) {
    int i = blockIdx.x * blockDim.x + threadIdx.x;
    if (i < NNODES) deg[i] = 1.0f;                 // self-loop weight
}
__global__ void k_deg_edges(const int* __restrict__ dst, const float* __restrict__ w,
                            float* deg) {
    int i = blockIdx.x * blockDim.x + threadIdx.x;
    if (i < NEDGES) atomicAdd(&deg[dst[i]], w[i]);
}
__global__ void k_dis(float* deg) {
    int i = blockIdx.x * blockDim.x + threadIdx.x;
    if (i < NNODES) {
        float d = deg[i];
        deg[i] = (d > 0.f) ? rsqrtf(fmaxf(d, BN_EPS)) : 0.f;
    }
}

// out[i,c] = dis[i]^2 * hlin[i,c] + bias[c]   (self-loop term + conv bias)
__global__ void k_gcn_init(const float* __restrict__ hlin, const float* __restrict__ dis,
                           const float* __restrict__ bias, float* __restrict__ out, int C) {
    int t = blockIdx.x * blockDim.x + threadIdx.x;
    if (t >= NNODES * C) return;
    int i = t / C, c = t - i * C;
    float di = dis[i];
    out[t] = di * di * hlin[t] + bias[c];
}

// per-edge scatter: out[dst] += dis[src]*w*dis[dst] * hlin[src]  (float4 chunks)
__global__ void k_gcn_edges(const int* __restrict__ src, const int* __restrict__ dst,
                            const float* __restrict__ w, const float* __restrict__ dis,
                            const float* __restrict__ hlin, float* __restrict__ out, int C) {
    int cpe = C >> 2;
    long t = (long)blockIdx.x * blockDim.x + threadIdx.x;
    if (t >= (long)NEDGES * cpe) return;
    int e = (int)(t / cpe);
    int c = ((int)(t - (long)e * cpe)) << 2;
    int s = src[e], d = dst[e];
    float nrm = dis[s] * w[e] * dis[d];
    const float4 hv = *reinterpret_cast<const float4*>(hlin + (size_t)s * C + c);
    float* o = out + (size_t)d * C + c;
    atomicAdd(o + 0, nrm * hv.x);
    atomicAdd(o + 1, nrm * hv.y);
    atomicAdd(o + 2, nrm * hv.z);
    atomicAdd(o + 3, nrm * hv.w);
}

// ---------------------------------------------------------------------------
// BatchNorm (training-mode, biased variance): one block per channel
// ---------------------------------------------------------------------------
__global__ void k_bn_stats(const float* __restrict__ x, float* __restrict__ mean,
                           float* __restrict__ rstd, int n, int C) {
    int c = blockIdx.x;
    float s = 0.f, s2 = 0.f;
    for (int i = threadIdx.x; i < n; i += blockDim.x) {
        __builtin_prefetch(&x[(size_t)(i + blockDim.x) * C + c], 0, 1);
        float v = x[(size_t)i * C + c];
        s += v; s2 += v * v;
    }
    __shared__ float sh[256], sh2[256];
    sh[threadIdx.x] = s; sh2[threadIdx.x] = s2;
    __syncthreads();
    for (int o = blockDim.x >> 1; o > 0; o >>= 1) {
        if ((int)threadIdx.x < o) {
            sh[threadIdx.x]  += sh[threadIdx.x + o];
            sh2[threadIdx.x] += sh2[threadIdx.x + o];
        }
        __syncthreads();
    }
    if (threadIdx.x == 0) {
        float mu = sh[0] / (float)n;
        float va = sh2[0] / (float)n - mu * mu;
        mean[c] = mu;
        rstd[c] = rsqrtf(va + BN_EPS);
    }
}

// out = relu(bn(x)) + shortcut (sc==nullptr -> no shortcut)
__global__ void k_bn_relu_add(const float* __restrict__ x, const float* __restrict__ mean,
                              const float* __restrict__ rstd, const float* __restrict__ g,
                              const float* __restrict__ be, const float* __restrict__ sc,
                              float* __restrict__ out, int n, int C) {
    int t = blockIdx.x * blockDim.x + threadIdx.x;
    if (t >= n * C) return;
    int c = t % C;
    float v = (x[t] - mean[c]) * rstd[c] * g[c] + be[c];
    v = fmaxf(v, 0.f);
    out[t] = v + (sc ? sc[t] : 0.f);
}

// ---------------------------------------------------------------------------
// Global mean/max pooling per graph
// ---------------------------------------------------------------------------
__device__ __forceinline__ void atomicMaxF(float* addr, float v) {
    if (v >= 0.f) atomicMax((int*)addr, __float_as_int(v));
    else          atomicMin((unsigned int*)addr, __float_as_uint(v));
}

__global__ void k_pool_init(float* p1, float* p2, float* cnt) {
    int t = blockIdx.x * blockDim.x + threadIdx.x;
    if (t >= NGRAPH * 128) return;
    p1[t] = 0.f;
    p2[t] = -__builtin_inff();
    if ((t & 127) == 0) cnt[t >> 7] = 0.f;
}

__global__ void k_pool_scatter(const float* __restrict__ h, const int* __restrict__ batch,
                               float* p1, float* p2, float* cnt) {
    long t = (long)blockIdx.x * blockDim.x + threadIdx.x;
    if (t >= (long)NNODES * 32) return;            // 128 channels / 4 per thread
    int i = (int)(t >> 5);
    int c = ((int)t & 31) << 2;
    int b = batch[i];
    float4 v = *reinterpret_cast<const float4*>(h + (size_t)i * 128 + c);
    float* a1 = p1 + (size_t)b * 128 + c;
    float* a2 = p2 + (size_t)b * 128 + c;
    atomicAdd(a1 + 0, v.x); atomicAdd(a1 + 1, v.y);
    atomicAdd(a1 + 2, v.z); atomicAdd(a1 + 3, v.w);
    atomicMaxF(a2 + 0, v.x); atomicMaxF(a2 + 1, v.y);
    atomicMaxF(a2 + 2, v.z); atomicMaxF(a2 + 3, v.w);
    if ((t & 31) == 0) atomicAdd(&cnt[b], 1.f);
}

__global__ void k_pool_finish(const float* p1, const float* p2, const float* cnt, float* z) {
    int t = blockIdx.x * blockDim.x + threadIdx.x;
    if (t >= NGRAPH * 128) return;
    int b = t >> 7, c = t & 127;
    z[(size_t)b * 256 + c]       = p1[t] / fmaxf(cnt[b], 1.f);
    z[(size_t)b * 256 + 128 + c] = p2[t];
}

// final linear 128 -> 1
__global__ void k_final(const float* __restrict__ y, const float* __restrict__ w,
                        const float* __restrict__ b, float* __restrict__ out) {
    int r = blockIdx.x * blockDim.x + threadIdx.x;
    if (r >= NGRAPH) return;
    float s = b[0];
    #pragma unroll 4
    for (int c = 0; c < 128; ++c) s += y[(size_t)r * 128 + c] * w[c];
    out[r] = s;
}

// ---------------------------------------------------------------------------
static inline unsigned nblk(long total, int bs) { return (unsigned)((total + bs - 1) / bs); }

extern "C" void kernel_launch(void* const* d_in, const int* in_sizes, int n_in,
                              void* d_out, int out_size, void* d_ws, size_t ws_size,
                              hipStream_t stream) {
    const float* x     = (const float*)d_in[0];
    const int*   ei    = (const int*)  d_in[1];
    const float* ew    = (const float*)d_in[2];
    const int*   batch = (const int*)  d_in[3];
    const float* W1  = (const float*)d_in[4];
    const float* b1  = (const float*)d_in[5];
    const float* g1  = (const float*)d_in[6];
    const float* be1 = (const float*)d_in[7];
    const float* sW1 = (const float*)d_in[8];
    const float* sb1 = (const float*)d_in[9];
    const float* W2  = (const float*)d_in[10];
    const float* b2  = (const float*)d_in[11];
    const float* g2  = (const float*)d_in[12];
    const float* be2 = (const float*)d_in[13];
    const float* W3  = (const float*)d_in[14];
    const float* b3  = (const float*)d_in[15];
    const float* g3  = (const float*)d_in[16];
    const float* be3 = (const float*)d_in[17];
    const float* sW3 = (const float*)d_in[18];
    const float* sb3 = (const float*)d_in[19];
    const float* fW1 = (const float*)d_in[20];
    const float* fb1 = (const float*)d_in[21];
    const float* fg  = (const float*)d_in[22];
    const float* fbe = (const float*)d_in[23];
    const float* fW2 = (const float*)d_in[24];
    const float* fb2 = (const float*)d_in[25];

    const int* src = ei;
    const int* dst = ei + NEDGES;

    // workspace carve-up (floats, all sub-regions 16B aligned)
    float* ws = (float*)d_ws;
    const size_t nb = (size_t)NNODES * 128;
    float* P    = ws;
    float* Q    = P + nb;
    float* R    = Q + nb;
    float* S    = R + nb;
    float* dis  = S + nb;                       // [N]
    float* mean = dis + NNODES;                 // [128]
    float* rstd = mean + 128;                   // [128]
    float* p1   = rstd + 128;                   // [B*128]
    float* p2   = p1 + (size_t)NGRAPH * 128;    // [B*128]
    float* cnt  = p2 + (size_t)NGRAPH * 128;    // [B]
    float* z    = cnt + NGRAPH;                 // [B*256]
    float* y1   = z + (size_t)NGRAPH * 256;     // [B*128]
    float* y2   = y1 + (size_t)NGRAPH * 128;    // [B*128]
    _Float16* a16 = (_Float16*)(y2 + (size_t)NGRAPH * 128); // [N*64] halves max
    _Float16* w16 = a16 + (size_t)NNODES * 64;  // [128*256] halves max

    const int GB = 128;  // 4 waves / block for GEMM

    // ---- symmetric GCN normalization (shared by all 3 layers) ----
    k_deg_init <<<nblk(NNODES, 256), 256, 0, stream>>>(dis);
    k_deg_edges<<<nblk(NEDGES, 256), 256, 0, stream>>>(dst, ew, dis);
    k_dis      <<<nblk(NNODES, 256), 256, 0, stream>>>(dis);

    // ---- block 1: GCN(4->64)+BN+ReLU, shortcut Linear(4->64) ----
    k_cvt_a <<<nblk((long)NNODES * 32, 256), 256, 0, stream>>>(x, a16, NNODES, 4, 32);
    k_cvt_wt<<<nblk(64 * 32, 256), 256, 0, stream>>>(W1, w16, 4, 64, 32);
    k_wmma_gemm<32, 4><<<nblk(NNODES / 16, 4), GB, 0, stream>>>(a16, w16, nullptr, P, NNODES, 0);
    k_cvt_wt<<<nblk(64 * 32, 256), 256, 0, stream>>>(sW1, w16, 4, 64, 32);
    k_wmma_gemm<32, 4><<<nblk(NNODES / 16, 4), GB, 0, stream>>>(a16, w16, sb1, Q, NNODES, 1);
    k_gcn_init <<<nblk((long)NNODES * 64, 256), 256, 0, stream>>>(P, dis, b1, R, 64);
    k_gcn_edges<<<nblk((long)NEDGES * 16, 256), 256, 0, stream>>>(src, dst, ew, dis, P, R, 64);
    k_bn_stats <<<64, 256, 0, stream>>>(R, mean, rstd, NNODES, 64);
    k_bn_relu_add<<<nblk((long)NNODES * 64, 256), 256, 0, stream>>>(R, mean, rstd, g1, be1, Q, S, NNODES, 64);

    // ---- block 2: GCN(64->64)+BN+ReLU, identity shortcut ----
    k_cvt_a <<<nblk((long)NNODES * 64, 256), 256, 0, stream>>>(S, a16, NNODES, 64, 64);
    k_cvt_wt<<<nblk(64 * 64, 256), 256, 0, stream>>>(W2, w16, 64, 64, 64);
    k_wmma_gemm<64, 4><<<nblk(NNODES / 16, 4), GB, 0, stream>>>(a16, w16, nullptr, P, NNODES, 0);
    k_gcn_init <<<nblk((long)NNODES * 64, 256), 256, 0, stream>>>(P, dis, b2, R, 64);
    k_gcn_edges<<<nblk((long)NEDGES * 16, 256), 256, 0, stream>>>(src, dst, ew, dis, P, R, 64);
    k_bn_stats <<<64, 256, 0, stream>>>(R, mean, rstd, NNODES, 64);
    k_bn_relu_add<<<nblk((long)NNODES * 64, 256), 256, 0, stream>>>(R, mean, rstd, g2, be2, S, Q, NNODES, 64);

    // ---- block 3: GCN(64->128)+BN+ReLU, shortcut Linear(64->128) ----
    k_cvt_a <<<nblk((long)NNODES * 64, 256), 256, 0, stream>>>(Q, a16, NNODES, 64, 64);
    k_cvt_wt<<<nblk(128 * 64, 256), 256, 0, stream>>>(W3, w16, 64, 128, 64);
    k_wmma_gemm<64, 8><<<nblk(NNODES / 16, 4), GB, 0, stream>>>(a16, w16, nullptr, P, NNODES, 0);
    k_cvt_wt<<<nblk(128 * 64, 256), 256, 0, stream>>>(sW3, w16, 64, 128, 64);
    k_wmma_gemm<64, 8><<<nblk(NNODES / 16, 4), GB, 0, stream>>>(a16, w16, sb3, S, NNODES, 1);
    k_gcn_init <<<nblk((long)NNODES * 128, 256), 256, 0, stream>>>(P, dis, b3, R, 128);
    k_gcn_edges<<<nblk((long)NEDGES * 32, 256), 256, 0, stream>>>(src, dst, ew, dis, P, R, 128);
    k_bn_stats <<<128, 256, 0, stream>>>(R, mean, rstd, NNODES, 128);
    k_bn_relu_add<<<nblk((long)NNODES * 128, 256), 256, 0, stream>>>(R, mean, rstd, g3, be3, S, P, NNODES, 128);
    // h3 = P (N x 128)

    // ---- global mean / max pooling ----
    k_pool_init   <<<nblk((long)NGRAPH * 128, 256), 256, 0, stream>>>(p1, p2, cnt);
    k_pool_scatter<<<nblk((long)NNODES * 32, 256), 256, 0, stream>>>(P, batch, p1, p2, cnt);
    k_pool_finish <<<nblk((long)NGRAPH * 128, 256), 256, 0, stream>>>(p1, p2, cnt, z);

    // ---- head: Linear(256->128)+BN+ReLU, Linear(128->1) ----
    k_cvt_a <<<nblk((long)NGRAPH * 256, 256), 256, 0, stream>>>(z, a16, NGRAPH, 256, 256);
    k_cvt_wt<<<nblk(128 * 256, 256), 256, 0, stream>>>(fW1, w16, 256, 128, 256);
    k_wmma_gemm<256, 8><<<nblk(NGRAPH / 16, 4), GB, 0, stream>>>(a16, w16, fb1, y1, NGRAPH, 1);
    k_bn_stats <<<128, 256, 0, stream>>>(y1, mean, rstd, NGRAPH, 128);
    k_bn_relu_add<<<nblk((long)NGRAPH * 128, 256), 256, 0, stream>>>(y1, mean, rstd, fg, fbe, nullptr, y2, NGRAPH, 128);
    k_final<<<nblk(NGRAPH, 256), 256, 0, stream>>>(y2, fW2, fb2, (float*)d_out);
}
